// PatientGraphSummary_34557306863749
// MI455X (gfx1250) — matrix operand order repaired and checked by reading, not compiled, not run on hardware
//
#include <hip/hip_runtime.h>
#include <hip/hip_bf16.h>

// ---------------------------------------------------------------------------
// PatientGraphSummary on MI455X (gfx1250, wave32, WMMA)
//
// out layout (floats): [0,P*D) summary | [P*D,2P*D) local | [2P*D,3P*D) global
// ws layout (floats):
//   accC  [0,        131072)  P*D f32 accumulator for maskf@nodes
//   denom [131072,   131584)  P
//   gsum  [131584,   131840)  D column-sum accumulator
//   g     [131840,   132096)  D mean
//   gvec  [132096,   132352)  D  (g @ W_gate_bot + b_gate)
//   gvec2 [132352,   132608)  D  (g @ W_global + b_global)
// total 132608 floats = 530,432 bytes of workspace.
// ---------------------------------------------------------------------------

#define NN 100000
#define DD 256
#define PP 512
#define ROWS 64            // P-rows per block in the main GEMM
#define KSPLIT 25          // 3125 chunks of 32 = 25 * 125 exactly
#define CHUNKS_PER 125
#define LDSTRIDE 40        // bf16 row stride: 80B, 16B aligned, bank-spread

typedef __attribute__((ext_vector_type(4)))  __bf16 v4bf;
typedef __attribute__((ext_vector_type(8)))  __bf16 v8bf;
typedef __attribute__((ext_vector_type(16))) __bf16 v16bf;
typedef __attribute__((ext_vector_type(8)))  float  v8f;

__device__ __forceinline__ v16bf cat16(v8bf a, v8bf b) {
  return __builtin_shufflevector(a, b, 0,1,2,3,4,5,6,7,8,9,10,11,12,13,14,15);
}

__device__ __forceinline__ v8f wmma_bf16(v16bf a, v16bf b, v8f c) {
  return __builtin_amdgcn_wmma_f32_16x16x32_bf16(
      /*neg_a=*/false, a, /*neg_b=*/false, b,
      /*c_mod=*/(short)0, c, /*reuse_a=*/false, /*reuse_b=*/false);
}

// ---- Kernel A: zero workspace ---------------------------------------------
__global__ void pgs_zero_ws(float* __restrict__ ws, int n) {
  int i = blockIdx.x * blockDim.x + threadIdx.x;
  if (i < n) ws[i] = 0.0f;
}

// ---- Kernel B: column-sum partials of nodes -> gsum -----------------------
__global__ void pgs_colsum(const float* __restrict__ nodes, float* __restrict__ gsum) {
  const int t = threadIdx.x;            // column, D == blockDim.x == 256
  const int r0 = blockIdx.x * 500;      // 200 blocks * 500 rows = 100000
  float s = 0.0f;
  for (int r = 0; r < 500; ++r)
    s += nodes[(r0 + r) * DD + t];
  atomicAdd(&gsum[t], s);
}

// ---- Kernel C: masked sum  accC += maskf[64 rows] @ nodes (bf16 WMMA) -----
// Grid (P/64, KSPLIT) x 256 threads. Double-buffered LDS, 4x2 tiles per wave.
__global__ void __launch_bounds__(256)
pgs_masked_mm(const float* __restrict__ nodes, const int* __restrict__ mask,
              float* __restrict__ accC, float* __restrict__ denomAcc) {
  __shared__ __bf16 ldsA[2][ROWS * LDSTRIDE];   // mask tile  [64 x 32] x2
  __shared__ __bf16 ldsB[2][DD * LDSTRIDE];     // nodes tile transposed [256 x 32] x2
  __shared__ float  denomLds[ROWS];

  const int tid   = threadIdx.x;
  const int lane  = tid & 31;
  const int wave  = tid >> 5;                 // 0..7, owns D cols [wave*32, +32)
  const int pBase = blockIdx.x * ROWS;
  const int ks    = blockIdx.y;               // K split 0..KSPLIT-1

  // mask staging: int4 per thread x2 -> rows (tid>>3) and (tid>>3)+32, k4=(tid&7)*4
  const int mRowS = tid >> 3;                 // 0..31
  const int mK4   = (tid & 7) * 4;

  if (tid < ROWS) denomLds[tid] = 0.0f;

  v8f acc[4][2] = {};                         // [rowTile][colTile]
  float rs0 = 0.0f, rs1 = 0.0f;

  const int h     = lane >> 4;                // K-half for A operand
  const int mrow  = lane & 15;                // M row within tile for A operand
  const int nCol  = lane & 15;                // N col for B operand
  const int khalf = lane >> 4;                // K-half for B operand
  const int dA    = wave * 32 + nCol;

  float nbuf[32];                             // staged nodes (registers)
  int4  mbuf[2];                              // staged mask  (registers)

  auto stage_load = [&](int ch) {
    const int kBase = (ks * CHUNKS_PER + ch) * 32;
    const int4* mask4 = (const int4*)mask;
#pragma unroll
    for (int j = 0; j < 2; ++j)
      mbuf[j] = mask4[(((pBase + mRowS + j * 32) * NN) + kBase + mK4) >> 2];
#pragma unroll
    for (int k = 0; k < 32; ++k)
      nbuf[k] = nodes[(kBase + k) * DD + tid];
  };
  auto stage_store = [&](int buf) {
#pragma unroll
    for (int j = 0; j < 2; ++j) {
      float m0 = (float)mbuf[j].x, m1 = (float)mbuf[j].y;
      float m2 = (float)mbuf[j].z, m3 = (float)mbuf[j].w;
      v4bf mv = {(__bf16)m0, (__bf16)m1, (__bf16)m2, (__bf16)m3};
      *(v4bf*)&ldsA[buf][(mRowS + j * 32) * LDSTRIDE + mK4] = mv;
      if (j == 0) rs0 += m0 + m1 + m2 + m3;
      else        rs1 += m0 + m1 + m2 + m3;
    }
#pragma unroll
    for (int k = 0; k < 32; ++k)
      ldsB[buf][tid * LDSTRIDE + k] = (__bf16)nbuf[k];
  };

  // prologue: stage chunk 0 into buffer 0
  stage_load(0);
  stage_store(0);

  for (int ch = 0; ch < CHUNKS_PER; ++ch) {
    const int cur = ch & 1;
    __syncthreads();   // buf[cur] staged & previous reads of buf[cur^1] done

    // operand reads for current buffer (DS loads)
    v16bf A[4];
#pragma unroll
    for (int rt = 0; rt < 4; ++rt) {
      const __bf16* ap = &ldsA[cur][(mrow + rt * 16) * LDSTRIDE + h * 16];
      A[rt] = cat16(*(const v8bf*)ap, *(const v8bf*)(ap + 8));
    }
    const __bf16* bp0 = &ldsB[cur][dA * LDSTRIDE + khalf * 16];
    const __bf16* bp1 = &ldsB[cur][(dA + 16) * LDSTRIDE + khalf * 16];
    v16bf B0 = cat16(*(const v8bf*)bp0, *(const v8bf*)(bp0 + 8));
    v16bf B1 = cat16(*(const v8bf*)bp1, *(const v8bf*)(bp1 + 8));

    // issue next chunk's global loads (latency hidden behind WMMAs)
    const bool more = (ch + 1) < CHUNKS_PER;
    if (more) stage_load(ch + 1);

#pragma unroll
    for (int rt = 0; rt < 4; ++rt) {
      acc[rt][0] = wmma_bf16(A[rt], B0, acc[rt][0]);
      acc[rt][1] = wmma_bf16(A[rt], B1, acc[rt][1]);
    }

    // convert + store next chunk into the other buffer
    if (more) stage_store(cur ^ 1);
  }

  // denominator reduction: thread holds partials for rows mRowS, mRowS+32
  atomicAdd(&denomLds[mRowS], rs0);
  atomicAdd(&denomLds[mRowS + 32], rs1);
  __syncthreads();
  if (tid < ROWS) atomicAdd(&denomAcc[pBase + tid], denomLds[tid]);

  // C epilogue: lane L -> N = L%16, rows M = r + (L/16)*8 within each row tile
#pragma unroll
  for (int rt = 0; rt < 4; ++rt) {
#pragma unroll
    for (int r = 0; r < 8; ++r) {
      int p = pBase + rt * 16 + r + (lane >> 4) * 8;
      atomicAdd(&accC[p * DD + dA],      acc[rt][0][r]);
      atomicAdd(&accC[p * DD + dA + 16], acc[rt][1][r]);
    }
  }
}

// ---- Kernel D: g = gsum/N ; gvec = g@Wg_bot + b_gate ; gvec2 = g@Wglob + b_global
__global__ void pgs_prep(const float* __restrict__ gsum,
                         const float* __restrict__ W_gate, const float* __restrict__ b_gate,
                         const float* __restrict__ W_global, const float* __restrict__ b_global,
                         float* __restrict__ g, float* __restrict__ gvec, float* __restrict__ gvec2) {
  __shared__ float gl[DD];
  const int d = threadIdx.x;
  float gv = gsum[d] * (1.0f / (float)NN);
  g[d] = gv;
  gl[d] = gv;
  __syncthreads();
  float s1 = b_gate[d], s2 = b_global[d];
  for (int k = 0; k < DD; ++k) {
    float gk = gl[k];
    s1 += gk * W_gate[(DD + k) * DD + d];   // bottom half of [2D, D]
    s2 += gk * W_global[k * DD + d];
  }
  gvec[d] = s1;
  gvec2[d] = s2;
}

// ---- Kernel E: local = accC / max(denom, 1) -> out1 ------------------------
__global__ void pgs_local_fin(const float* __restrict__ accC, const float* __restrict__ denom,
                              float* __restrict__ out1) {
  int i = blockIdx.x * blockDim.x + threadIdx.x;   // P*D elements
  int p = i >> 8;
  out1[i] = accC[i] / fmaxf(denom[p], 1.0f);
}

// ---- Kernel F: gate GEMMs + sigmoid blend (bf16 WMMA) ----------------------
__global__ void __launch_bounds__(256)
pgs_gate(const float* __restrict__ local, const float* __restrict__ W_gate,
         const float* __restrict__ W_local, const float* __restrict__ b_local,
         const float* __restrict__ g, const float* __restrict__ gvec,
         const float* __restrict__ gvec2,
         float* __restrict__ out0, float* __restrict__ out2) {
  __shared__ __bf16 ldsA [16 * LDSTRIDE];     // local tile [16 x 32]
  __shared__ __bf16 ldsB1[DD * LDSTRIDE];     // W_gate_top tile transposed
  __shared__ __bf16 ldsB2[DD * LDSTRIDE];     // W_local tile transposed

  const int tid   = threadIdx.x;
  const int lane  = tid & 31;
  const int wave  = tid >> 5;
  const int pBase = blockIdx.x * 16;

  const int rowS = tid >> 5;
  const int kS   = tid & 31;
  const int h     = lane >> 4;
  const int mrow  = lane & 15;
  const int nCol  = lane & 15;
  const int khalf = lane >> 4;
  const int d0    = wave * 32 + nCol;

  v8f cg0 = {}; v8f cg1 = {}; v8f cl0 = {}; v8f cl1 = {};

  for (int ch = 0; ch < 8; ++ch) {            // K = 256 = 8 chunks of 32
    const int kBase = ch * 32;

    ldsA[rowS * LDSTRIDE + kS]       = (__bf16)local[(pBase + rowS)     * DD + kBase + kS];
    ldsA[(rowS + 8) * LDSTRIDE + kS] = (__bf16)local[(pBase + rowS + 8) * DD + kBase + kS];
#pragma unroll 8
    for (int k = 0; k < 32; ++k) {
      ldsB1[tid * LDSTRIDE + k] = (__bf16)W_gate [(kBase + k) * DD + tid]; // top half
      ldsB2[tid * LDSTRIDE + k] = (__bf16)W_local[(kBase + k) * DD + tid];
    }
    __syncthreads();

    v16bf A = cat16(*(const v8bf*)&ldsA[mrow * LDSTRIDE + h * 16],
                    *(const v8bf*)&ldsA[mrow * LDSTRIDE + 8 + h * 16]);

    const __bf16* g0 = &ldsB1[d0 * LDSTRIDE + khalf * 16];
    const __bf16* g1 = &ldsB1[(d0 + 16) * LDSTRIDE + khalf * 16];
    const __bf16* l0 = &ldsB2[d0 * LDSTRIDE + khalf * 16];
    const __bf16* l1 = &ldsB2[(d0 + 16) * LDSTRIDE + khalf * 16];

    cg0 = wmma_bf16(A, cat16(*(const v8bf*)g0, *(const v8bf*)(g0 + 8)), cg0);
    cg1 = wmma_bf16(A, cat16(*(const v8bf*)g1, *(const v8bf*)(g1 + 8)), cg1);
    cl0 = wmma_bf16(A, cat16(*(const v8bf*)l0, *(const v8bf*)(l0 + 8)), cl0);
    cl1 = wmma_bf16(A, cat16(*(const v8bf*)l1, *(const v8bf*)(l1 + 8)), cl1);

    __syncthreads();
  }

  // epilogue: summary + global_summary broadcast
#pragma unroll
  for (int r = 0; r < 8; ++r) {
    int p = pBase + r + (lane >> 4) * 8;
#pragma unroll
    for (int t = 0; t < 2; ++t) {
      int d = d0 + t * 16;
      float ga = (t == 0 ? cg0[r] : cg1[r]) + gvec[d];
      float la = (t == 0 ? cl0[r] : cl1[r]) + b_local[d];
      float gate = 1.0f / (1.0f + __expf(-ga));
      out0[p * DD + d] = gate * la + (1.0f - gate) * gvec2[d];
      out2[p * DD + d] = g[d];
    }
  }
}

extern "C" void kernel_launch(void* const* d_in, const int* in_sizes, int n_in,
                              void* d_out, int out_size, void* d_ws, size_t ws_size,
                              hipStream_t stream) {
  (void)in_sizes; (void)n_in; (void)out_size; (void)ws_size;
  const float* nodes    = (const float*)d_in[0];
  const int*   mask     = (const int*)  d_in[1];
  const float* W_local  = (const float*)d_in[2];
  const float* b_local  = (const float*)d_in[3];
  const float* W_global = (const float*)d_in[4];
  const float* b_global = (const float*)d_in[5];
  const float* W_gate   = (const float*)d_in[6];
  const float* b_gate   = (const float*)d_in[7];

  float* out  = (float*)d_out;
  float* out0 = out;                 // summary
  float* out1 = out + PP * DD;       // local_summary
  float* out2 = out + 2 * PP * DD;   // global_summary

  float* ws    = (float*)d_ws;
  float* accC  = ws;                 // P*D
  float* denom = ws + PP * DD;       // P
  float* gsum  = denom + PP;         // D
  float* g     = gsum + DD;          // D
  float* gvec  = g + DD;             // D
  float* gvec2 = gvec + DD;          // D
  const int wsFloats = PP * DD + PP + 4 * DD;   // 132608

  pgs_zero_ws<<<(wsFloats + 255) / 256, 256, 0, stream>>>(ws, wsFloats);
  pgs_colsum<<<200, 256, 0, stream>>>(nodes, gsum);
  dim3 gridC(PP / ROWS, KSPLIT);
  pgs_masked_mm<<<gridC, 256, 0, stream>>>(nodes, mask, accC, denom);
  pgs_prep<<<1, 256, 0, stream>>>(gsum, W_gate, b_gate, W_global, b_global, g, gvec, gvec2);
  pgs_local_fin<<<(PP * DD) / 256, 256, 0, stream>>>(accC, denom, out1);
  pgs_gate<<<PP / 16, 256, 0, stream>>>(out1, W_gate, W_local, b_local, g, gvec, gvec2, out0, out2);
}
